// MicroTransformer_49100066127918
// MI455X (gfx1250) — matrix operand order repaired and checked by reading, not compiled, not run on hardware
//
#include <hip/hip_runtime.h>
#include <hip/hip_bf16.h>
#include <math.h>

// ---------------------------------------------------------------------------
// MicroTransformer forward for MI455X (gfx1250, wave32, WMMA).
// All GEMMs: v_wmma_f32_16x16x32_f16, f32 accumulate, f16 operands staged in
// LDS via GLOBAL_LOAD_ASYNC_TO_LDS_B128 when the toolchain exposes it.
// ---------------------------------------------------------------------------

typedef __attribute__((ext_vector_type(16))) _Float16 v16h;
typedef __attribute__((ext_vector_type(8)))  _Float16 v8h;
typedef __attribute__((ext_vector_type(4)))  _Float16 v4h;
typedef __attribute__((ext_vector_type(8)))  float    v8f;
typedef int v4i __attribute__((vector_size(16)));

#define MT_B      2
#define MT_T      2048
#define MT_M      4096      // B*T
#define MT_D      512
#define MT_H      8
#define MT_HD     64
#define MT_HF     2048
#define MT_V      8000
#define MT_L      4

union V16U { v16h v; v8h h[2]; };

// ---- gfx1250 async LDS copy path (guarded; falls back to b128 ld/st) ------
#if defined(__HIP_DEVICE_COMPILE__) && __has_builtin(__builtin_amdgcn_global_load_async_to_lds_b128)
#define MT_ASYNC_LDS 1
#else
#define MT_ASYNC_LDS 0
#endif

__device__ __forceinline__ void mt_cp128(const _Float16* g, _Float16* l) {
#if MT_ASYNC_LDS
    auto gp = (__attribute__((address_space(1))) v4i*)(unsigned long long)(uintptr_t)g;
    auto lp = (__attribute__((address_space(3))) v4i*)(unsigned)(uintptr_t)l;
    __builtin_amdgcn_global_load_async_to_lds_b128(gp, lp, 0, 0);
#else
    *(v8h*)l = *(const v8h*)g;
#endif
}

__device__ __forceinline__ void mt_wait_async() {
#if MT_ASYNC_LDS
#if __has_builtin(__builtin_amdgcn_s_wait_asynccnt)
    __builtin_amdgcn_s_wait_asynccnt(0);
#else
    asm volatile("s_wait_asynccnt 0" ::: "memory");
#endif
#endif
}

// ---------------------------------------------------------------------------
// f32 -> f16 bulk convert (weights, once per call); 4 elements / thread
// ---------------------------------------------------------------------------
__global__ void mt_cvt_f16(const float* __restrict__ in, _Float16* __restrict__ out) {
    int idx = (blockIdx.x * 256 + threadIdx.x) * 4;
    float4 v = *(const float4*)(in + idx);
    v4h o;
    o[0] = (_Float16)v.x; o[1] = (_Float16)v.y;
    o[2] = (_Float16)v.z; o[3] = (_Float16)v.w;
    *(v4h*)(out + idx) = o;
}

// ---------------------------------------------------------------------------
// Embedding gather: x[m, d] = emb[ids[m], d]
// ---------------------------------------------------------------------------
__global__ void mt_embed(const int* __restrict__ ids,
                         const float* __restrict__ emb,
                         float* __restrict__ x) {
    int idx = blockIdx.x * 256 + threadIdx.x;        // [0, M*D)
    int m = idx >> 9;                                // /512
    int d = idx & 511;
    x[idx] = emb[(size_t)ids[m] * MT_D + d];
}

// ---------------------------------------------------------------------------
// RMSNorm, one block (256 thr) per row of 512; writes f16 (GEMM A operand)
// ---------------------------------------------------------------------------
__global__ void mt_rmsnorm(const float* __restrict__ x,
                           const float* __restrict__ w,
                           _Float16* __restrict__ out) {
    int row = blockIdx.x;
    int tid = threadIdx.x;
    const float* xr = x + (size_t)row * MT_D;
    float a = xr[tid];
    float b = xr[tid + 256];
    float ss = a * a + b * b;
    #pragma unroll
    for (int off = 16; off >= 1; off >>= 1) ss += __shfl_xor(ss, off, 32);
    __shared__ float red[8];
    int wv = tid >> 5;
    if ((tid & 31) == 0) red[wv] = ss;
    __syncthreads();
    float tot = 0.f;
    #pragma unroll
    for (int i = 0; i < 8; ++i) tot += red[i];
    float inv = rsqrtf(tot * (1.0f / MT_D) + 1e-6f);
    _Float16* orow = out + (size_t)row * MT_D;
    orow[tid]       = (_Float16)(a * inv * w[tid]);
    orow[tid + 256] = (_Float16)(b * inv * w[tid + 256]);
}

// ---------------------------------------------------------------------------
// Tiled WMMA GEMM:  C[M,N] = A_f16[M,K] @ W_f16[N,K]^T  (+ resid)
// Block: 256 threads (8 wave32s), tile 128(M) x 64(N), BK = 32.
// Wave (wm,wn) owns a 32x32 C sub-tile: 2 A-frags x 2 B-frags -> 4 WMMAs.
// Both operands staged via async b128 copies to LDS.
// ---------------------------------------------------------------------------
template <bool RESID>
__global__ __launch_bounds__(256)
void mt_gemm(const _Float16* __restrict__ A,
             const _Float16* __restrict__ W,
             const float* __restrict__ resid,
             float* __restrict__ C,
             int M, int N, int K) {
    __shared__ _Float16 As[128 * 32];   // [m][k]  8 KB
    __shared__ _Float16 Bs[64 * 32];    // [n][k]  4 KB

    int tid  = threadIdx.x;
    int lane = tid & 31;
    int wv   = tid >> 5;
    int wm   = wv & 3;          // 0..3 -> 32-row M sub-tile
    int wn   = wv >> 2;         // 0..1 -> 32-col N sub-tile
    int bn   = blockIdx.x * 64;
    int bm   = blockIdx.y * 128;

    // staging coords: A = 16 halfs / thread (2 chunks), B = 8 halfs / thread
    int idxA = tid * 16;
    int arow = idxA >> 5;       // 0..127
    int acol = idxA & 31;       // 0 or 16
    int idxB = tid * 8;
    int brow = idxB >> 5;       // 0..63
    int bcol = idxB & 31;

    // fragment coords (ISA 7.12.2 16-bit layouts -> contiguous 16B chunks)
    int halfsel = lane >> 4;
    int mr0 = wm * 32 + (lane & 15);
    int nc0 = wn * 32 + (lane & 15);

    v8f c00 = {}, c01 = {}, c10 = {}, c11 = {};

    for (int k0 = 0; k0 < K; k0 += 32) {
        // ---- stage A tile ----
        const _Float16* ag = A + (size_t)(bm + arow) * K + k0 + acol;
        mt_cp128(ag,     As + idxA);
        mt_cp128(ag + 8, As + idxA + 8);
        // ---- stage W tile ----
        const _Float16* wg = W + (size_t)(bn + brow) * K + k0 + bcol;
        mt_cp128(wg, Bs + idxB);
        if (k0 + 32 < K) {
            __builtin_prefetch(ag + 32, 0, 1);
            __builtin_prefetch(wg + 32, 0, 1);
        }
        mt_wait_async();
        __syncthreads();

        // ---- fragment loads: contiguous 16B chunks from LDS ----
        const v8h* ar0 = (const v8h*)(As + mr0 * 32);
        const v8h* ar1 = (const v8h*)(As + (mr0 + 16) * 32);
        V16U a0, a1;
        a0.h[0] = ar0[halfsel];  a0.h[1] = ar0[2 + halfsel];
        a1.h[0] = ar1[halfsel];  a1.h[1] = ar1[2 + halfsel];

        const v8h* br0 = (const v8h*)(Bs + nc0 * 32);
        const v8h* br1 = (const v8h*)(Bs + (nc0 + 16) * 32);
        V16U b0, b1;
        b0.h[0] = br0[halfsel];  b0.h[1] = br0[2 + halfsel];
        b1.h[0] = br1[halfsel];  b1.h[1] = br1[2 + halfsel];

        c00 = __builtin_amdgcn_wmma_f32_16x16x32_f16(
            false, a0.v, false, b0.v, (short)0, c00, false, false);
        c01 = __builtin_amdgcn_wmma_f32_16x16x32_f16(
            false, a0.v, false, b1.v, (short)0, c01, false, false);
        c10 = __builtin_amdgcn_wmma_f32_16x16x32_f16(
            false, a1.v, false, b0.v, (short)0, c10, false, false);
        c11 = __builtin_amdgcn_wmma_f32_16x16x32_f16(
            false, a1.v, false, b1.v, (short)0, c11, false, false);

        __syncthreads();
    }

    // ---- store (C/D layout: VGPR r -> row (lane>=16 ? 8 : 0)+r, col lane&15)
    int m0 = bm + wm * 32 + halfsel * 8;
    int n0 = bn + wn * 32 + (lane & 15);
    #pragma unroll
    for (int r = 0; r < 8; ++r) {
        size_t o0 = (size_t)(m0 + r) * N + n0;
        size_t o1 = (size_t)(m0 + 16 + r) * N + n0;
        float v00 = c00[r], v01 = c01[r], v10 = c10[r], v11 = c11[r];
        if (RESID) {
            v00 += resid[o0]; v01 += resid[o0 + 16];
            v10 += resid[o1]; v11 += resid[o1 + 16];
        }
        C[o0]      = v00;  C[o0 + 16] = v01;
        C[o1]      = v10;  C[o1 + 16] = v11;
    }
}

// ---------------------------------------------------------------------------
// RoPE applied in-place on qkv buffer [M, 3*D]; one block per token row.
// ---------------------------------------------------------------------------
__global__ void mt_rope(float* __restrict__ qkv) {
    int m  = blockIdx.x;
    int t  = m & (MT_T - 1);
    int h  = threadIdx.x >> 5;
    int d2 = threadIdx.x & 31;
    float freq = __powf(10000.0f, -(float)(2 * d2) * (1.0f / MT_HD));
    float ang  = (float)t * freq;
    float c = __cosf(ang), s = __sinf(ang);
    float* q = qkv + (size_t)m * (3 * MT_D) + h * MT_HD;
    float* k = q + MT_D;
    float q0 = q[d2], q1 = q[d2 + 32];
    q[d2]      = q0 * c - q1 * s;
    q[d2 + 32] = q1 * c + q0 * s;
    float k0 = k[d2], k1 = k[d2 + 32];
    k[d2]      = k0 * c - k1 * s;
    k[d2 + 32] = k1 * c + k0 * s;
}

// ---------------------------------------------------------------------------
// Causal attention, online softmax. One wave per query; K/V tiles in LDS.
// ---------------------------------------------------------------------------
__global__ __launch_bounds__(256)
void mt_attn(const float* __restrict__ qkv, _Float16* __restrict__ out) {
    __shared__ float Ks[32][64];
    __shared__ float Vs[32][64];

    int blk    = blockIdx.x;
    int qchunk = blk & (MT_T / 8 - 1);
    int bh     = blk >> 8;
    int b      = bh >> 3;
    int h      = bh & 7;

    int tid  = threadIdx.x;
    int lane = tid & 31;
    int wv   = tid >> 5;
    int t    = qchunk * 8 + wv;
    int m    = b * MT_T + t;
    int d0   = 2 * lane;

    const float* qp = qkv + (size_t)m * (3 * MT_D) + h * MT_HD;
    float qa = qp[d0] * 0.125f;
    float qb = qp[d0 + 1] * 0.125f;

    float mrun = -3.0e38f, lrun = 0.f, acc0 = 0.f, acc1 = 0.f;
    int tmax = qchunk * 8 + 7;

    int sidx = tid * 8;
    int sr = sidx >> 6;
    int sc = sidx & 63;

    for (int kt0 = 0; kt0 <= tmax; kt0 += 32) {
        __syncthreads();
        const float* kp = qkv + (size_t)(b * MT_T + kt0 + sr) * (3 * MT_D)
                          + MT_D + h * MT_HD + sc;
        const float* vp = kp + MT_D;
        #pragma unroll
        for (int i = 0; i < 8; ++i) Ks[sr][sc + i] = kp[i];
        #pragma unroll
        for (int i = 0; i < 8; ++i) Vs[sr][sc + i] = vp[i];
        __syncthreads();

        int jmax = t - kt0; if (jmax > 31) jmax = 31;
        for (int j = 0; j <= jmax; ++j) {
            float s = qa * Ks[j][d0] + qb * Ks[j][d0 + 1];
            #pragma unroll
            for (int off = 16; off >= 1; off >>= 1) s += __shfl_xor(s, off, 32);
            float mn = fmaxf(mrun, s);
            float sc0 = __expf(mrun - mn);
            float p   = __expf(s - mn);
            lrun = lrun * sc0 + p;
            acc0 = acc0 * sc0 + p * Vs[j][d0];
            acc1 = acc1 * sc0 + p * Vs[j][d0 + 1];
            mrun = mn;
        }
    }

    float inv = 1.0f / lrun;
    _Float16* op = out + (size_t)m * MT_D + h * MT_HD;
    op[d0]     = (_Float16)(acc0 * inv);
    op[d0 + 1] = (_Float16)(acc1 * inv);
}

// ---------------------------------------------------------------------------
// SwiGLU gate: g = silu(a) * c, emitted as f16 (feeds w2 WMMA GEMM)
// ---------------------------------------------------------------------------
__global__ void mt_swiglu(const float* __restrict__ a,
                          const float* __restrict__ c,
                          _Float16* __restrict__ g) {
    int idx = blockIdx.x * 256 + threadIdx.x;
    float av = a[idx];
    float sv = av / (1.0f + __expf(-av));
    g[idx] = (_Float16)(sv * c[idx]);
}

// ---------------------------------------------------------------------------
// Host-side orchestration
// ---------------------------------------------------------------------------
extern "C" void kernel_launch(void* const* d_in, const int* in_sizes, int n_in,
                              void* d_out, int out_size, void* d_ws, size_t ws_size,
                              hipStream_t stream) {
    const int*   ids    = (const int*)d_in[0];
    const float* emb    = (const float*)d_in[1];
    const float* qkv_w  = (const float*)d_in[2];
    const float* proj_w = (const float*)d_in[3];
    const float* w1     = (const float*)d_in[4];
    const float* w2     = (const float*)d_in[5];
    const float* w3     = (const float*)d_in[6];
    const float* ln1    = (const float*)d_in[7];
    const float* ln2    = (const float*)d_in[8];
    const float* lnf    = (const float*)d_in[9];
    float* logits = (float*)d_out;

    char* ws = (char*)d_ws;
    size_t off = 0;
    auto alloc = [&](size_t bytes) -> void* {
        void* p = (void*)(ws + off);
        off += (bytes + 255) & ~(size_t)255;
        return p;
    };

    const size_t nQKV = (size_t)MT_L * 3 * MT_D * MT_D;   // 3,145,728
    const size_t nPRJ = (size_t)MT_L * MT_D * MT_D;       // 1,048,576
    const size_t nFF  = (size_t)MT_L * MT_HF * MT_D;      // 4,194,304
    const size_t nEMB = (size_t)MT_V * MT_D;              // 4,096,000

    float*     x      = (float*)    alloc((size_t)MT_M * MT_D * 4);
    _Float16*  hbuf   = (_Float16*) alloc((size_t)MT_M * MT_D * 2);
    float*     qkvb   = (float*)    alloc((size_t)MT_M * 3 * MT_D * 4);
    _Float16*  attnf  = (_Float16*) alloc((size_t)MT_M * MT_D * 2);
    float*     ffa    = (float*)    alloc((size_t)MT_M * MT_HF * 4);
    float*     ffc    = (float*)    alloc((size_t)MT_M * MT_HF * 4);
    _Float16*  gf16   = (_Float16*) alloc((size_t)MT_M * MT_HF * 2);
    _Float16*  qkv16  = (_Float16*) alloc(nQKV * 2);
    _Float16*  proj16 = (_Float16*) alloc(nPRJ * 2);
    _Float16*  w116   = (_Float16*) alloc(nFF * 2);
    _Float16*  w216   = (_Float16*) alloc(nFF * 2);
    _Float16*  w316   = (_Float16*) alloc(nFF * 2);
    _Float16*  emb16  = (_Float16*) alloc(nEMB * 2);

    dim3 blk(256);

    // one-shot weight conversion f32 -> f16 (halves weight traffic for GEMMs)
    mt_cvt_f16<<<dim3(nQKV / 1024), blk, 0, stream>>>(qkv_w,  qkv16);
    mt_cvt_f16<<<dim3(nPRJ / 1024), blk, 0, stream>>>(proj_w, proj16);
    mt_cvt_f16<<<dim3(nFF  / 1024), blk, 0, stream>>>(w1, w116);
    mt_cvt_f16<<<dim3(nFF  / 1024), blk, 0, stream>>>(w2, w216);
    mt_cvt_f16<<<dim3(nFF  / 1024), blk, 0, stream>>>(w3, w316);
    mt_cvt_f16<<<dim3(nEMB / 1024), blk, 0, stream>>>(emb, emb16);

    mt_embed<<<dim3(MT_M * MT_D / 256), blk, 0, stream>>>(ids, emb, x);

    for (int l = 0; l < MT_L; ++l) {
        // --- attention block ---
        mt_rmsnorm<<<dim3(MT_M), blk, 0, stream>>>(x, ln1 + l * MT_D, hbuf);
        mt_gemm<false><<<dim3(3 * MT_D / 64, MT_M / 128), blk, 0, stream>>>(
            hbuf, qkv16 + (size_t)l * 3 * MT_D * MT_D, nullptr, qkvb,
            MT_M, 3 * MT_D, MT_D);
        mt_rope<<<dim3(MT_M), blk, 0, stream>>>(qkvb);
        mt_attn<<<dim3(MT_B * MT_H * (MT_T / 8)), blk, 0, stream>>>(qkvb, attnf);
        mt_gemm<true><<<dim3(MT_D / 64, MT_M / 128), blk, 0, stream>>>(
            attnf, proj16 + (size_t)l * MT_D * MT_D, x, x,
            MT_M, MT_D, MT_D);

        // --- FFN block ---
        mt_rmsnorm<<<dim3(MT_M), blk, 0, stream>>>(x, ln2 + l * MT_D, hbuf);
        mt_gemm<false><<<dim3(MT_HF / 64, MT_M / 128), blk, 0, stream>>>(
            hbuf, w116 + (size_t)l * MT_HF * MT_D, nullptr, ffa,
            MT_M, MT_HF, MT_D);
        mt_gemm<false><<<dim3(MT_HF / 64, MT_M / 128), blk, 0, stream>>>(
            hbuf, w316 + (size_t)l * MT_HF * MT_D, nullptr, ffc,
            MT_M, MT_HF, MT_D);
        mt_swiglu<<<dim3(MT_M * MT_HF / 256), blk, 0, stream>>>(ffa, ffc, gf16);
        mt_gemm<true><<<dim3(MT_D / 64, MT_M / 128), blk, 0, stream>>>(
            gf16, w216 + (size_t)l * MT_D * MT_HF, x, x,
            MT_M, MT_D, MT_HF);
    }

    // --- final norm + tied lm_head ---
    mt_rmsnorm<<<dim3(MT_M), blk, 0, stream>>>(x, lnf, hbuf);
    mt_gemm<false><<<dim3(MT_V / 64, MT_M / 128), blk, 0, stream>>>(
        hbuf, emb16, nullptr, logits, MT_M, MT_V, MT_D);

    (void)in_sizes; (void)n_in; (void)out_size; (void)ws_size;
}
